// AttentionSampleUpdater_60301340835948
// MI455X (gfx1250) — matrix operand order repaired and checked by reading, not compiled, not run on hardware
//
#include <hip/hip_runtime.h>
#include <hip/hip_bf16.h>
#include <stdint.h>

// ---------------------------------------------------------------------------
// AttentionSampleUpdater for MI455X (gfx1250, wave32, WMMA + TDM + async-LDS)
//
//   N = 8192 nodes, D = 64 features, K = 16 samples, DEG = 32, C = 33*16 = 528
//
// One 256-thread workgroup per node:
//   Phase A: adj-row scan (the 256MB HBM-bound part) via
//            global_load_async_to_lds_b128 double-buffering (ASYNCcnt), each
//            thread owning a private 16B LDS slot (no barriers needed).
//   Phase B: gather 528 candidate sample indices into LDS.
//   Phase C: bitonic sort (padded to 1024 with INT_MAX).
//   Phase D: sims x[i]·x[cand] via v_wmma_f32_16x16x32_f16. The 16 candidate
//            feature rows per tile are fetched with TENSOR_LOAD_TO_LDS in
//            gather mode (sorted indices satisfy the TDM monotonicity rule),
//            double-buffered per wave with s_wait_tensorcnt <= 1.
//   Phase E: dedup mask (-1e30), iterative argmax top-16 (lower index wins
//            ties, matching jax top_k), <K-unique fallback path.
// ---------------------------------------------------------------------------

typedef __attribute__((ext_vector_type(16))) _Float16 v16h;
typedef __attribute__((ext_vector_type(8)))  float    v8f;
typedef __attribute__((ext_vector_type(4)))  unsigned v4u;
typedef __attribute__((ext_vector_type(8)))  int      v8i;
typedef __attribute__((ext_vector_type(4)))  int      v4i;

#define AS1 __attribute__((address_space(1)))
#define AS3 __attribute__((address_space(3)))

#define NODES_DEG   32
#define NODES_K     16
#define NODES_D     64
#define CAND_C      528          // 33 * 16
#define NTILES      33
#define SORT_N      1024
#define NEG_MASK    (-1.0e30f)
#define NEG_DEAD    (-3.0e38f)

#if __has_builtin(__builtin_amdgcn_global_load_async_to_lds_b128)
#define HAVE_ASYNC_LDS 1
#else
#define HAVE_ASYNC_LDS 0
#endif

#if __has_builtin(__builtin_amdgcn_tensor_load_to_lds)
#define HAVE_TDM 1
#else
#define HAVE_TDM 0
#endif

#if __has_builtin(__builtin_amdgcn_s_wait_asynccnt)
#define WAIT_ASYNC(N) do { __builtin_amdgcn_s_wait_asynccnt((short)(N)); \
                           asm volatile("" ::: "memory"); } while (0)
#else
#define WAIT_ASYNC(N) asm volatile("s_wait_asynccnt " #N ::: "memory")
#endif

#if __has_builtin(__builtin_amdgcn_s_wait_tensorcnt)
#define WAIT_TENSOR(N) do { __builtin_amdgcn_s_wait_tensorcnt((short)(N)); \
                            asm volatile("" ::: "memory"); } while (0)
#else
#define WAIT_TENSOR(N) asm volatile("s_wait_tensorcnt " #N ::: "memory")
#endif

__global__ void attn_cvt_f16_kernel(const float* __restrict__ x,
                                    _Float16* __restrict__ xh, int total) {
    int idx = blockIdx.x * blockDim.x + threadIdx.x;
    if (idx < total) xh[idx] = (_Float16)x[idx];
}

#if HAVE_TDM
// Issue one TDM gather of 16 rows (128B each) of xh into LDS at lds_addr.
// Descriptor per CDNA5 ISA (gather mode, 16-bit idx).
__device__ __forceinline__ void tdm_gather_rows(const _Float16* base,
                                                const int* rows,
                                                unsigned lds_addr, int n) {
    unsigned long long ga = (unsigned long long)(size_t)base;
    v4u g0;
    g0[0] = 1u | (1u << 31);                       // count=1, gather_mode=1, idx16
    g0[1] = lds_addr;                              // LDS byte address
    g0[2] = (unsigned)ga;                          // global_addr[31:0]
    g0[3] = ((unsigned)(ga >> 32) & 0x01FFFFFFu)   // global_addr[56:32]
            | (2u << 30);                          // type = 2 ("image")
    v8i g1;
    g1[0] = 0x00010000;                            // wg_mask=0, data_size=2B
    g1[1] = (int)(64u << 16);                      // tensor_dim0 = 64
    g1[2] = (int)(((unsigned)n & 0xFFFFu) << 16);  // tensor_dim1[15:0]
    g1[3] = (int)((((unsigned)n >> 16) & 0xFFFFu)  // tensor_dim1[31:16]
            | (64u << 16));                        // tile_dim0 = 64
    g1[4] = 16;                                    // tile_dim1 = #indices
    g1[5] = 64;                                    // tensor_dim0_stride = 64
    g1[6] = 0;
    g1[7] = 0;
    v4i g2, g3;
    g2[0] = (rows[0]  & 0xFFFF) | (rows[1]  << 16);
    g2[1] = (rows[2]  & 0xFFFF) | (rows[3]  << 16);
    g2[2] = (rows[4]  & 0xFFFF) | (rows[5]  << 16);
    g2[3] = (rows[6]  & 0xFFFF) | (rows[7]  << 16);
    g3[0] = (rows[8]  & 0xFFFF) | (rows[9]  << 16);
    g3[1] = (rows[10] & 0xFFFF) | (rows[11] << 16);
    g3[2] = (rows[12] & 0xFFFF) | (rows[13] << 16);
    g3[3] = (rows[14] & 0xFFFF) | (rows[15] << 16);
#if defined(__clang_major__) && __clang_major__ >= 23
    v8i z8 = {};
    __builtin_amdgcn_tensor_load_to_lds(g0, g1, g2, g3, z8, 0);
#else
    __builtin_amdgcn_tensor_load_to_lds(g0, g1, g2, g3, 0);
#endif
}
#endif // HAVE_TDM

__global__ __launch_bounds__(256)
void attn_sample_update_kernel(const float* __restrict__ adj,
                               const _Float16* __restrict__ xh,
                               const int* __restrict__ cs,
                               int* __restrict__ out, int n) {
    const int i = blockIdx.x;        // node id
    const int t = threadIdx.x;       // 0..255
    const int lane = t & 31;         // wave32
    const int wave = t >> 5;         // 8 waves

    __shared__ int      s_cand[SORT_N];
    __shared__ float    s_sim[CAND_C + 16];
    __shared__ _Float16 s_xi[NODES_D];
    __shared__ int      s_nbr[NODES_DEG];
    __shared__ int      s_sel[NODES_K];
    __shared__ int      s_uniq[NODES_K];
    __shared__ float    s_rval[256];
    __shared__ int      s_ridx[256];
    __shared__ int      s_cnt;
    __shared__ int      s_nu;
#if HAVE_ASYNC_LDS
    __shared__ alignas(16) float s_adj[2][1024];          // 8 KB double buffer
#endif
#if HAVE_TDM
    __shared__ alignas(32) _Float16 s_feat[8][2][16][NODES_D]; // 32 KB
#endif

    if (t == 0) { s_cnt = 0; s_nu = 0; }
    if (t < NODES_D) s_xi[t] = xh[(size_t)i * NODES_D + t];

    // Speculative prefetch of this block's adj row (global_prefetch_b8).
    __builtin_prefetch(adj + (size_t)i * n + t * (n / 256), 0, 0);
    __syncthreads();

    // ---------------- Phase A: neighbor extraction (HBM-bound) -------------
    {
#if HAVE_ASYNC_LDS
        // Async global->LDS double-buffered scan. Thread t owns the 16-byte
        // slot t of each chunk buffer; per-wave ASYNCcnt ordering is enough.
        const size_t rowbase = (size_t)i * n;
        const int nchunks = n >> 10;                       // 1024 floats/chunk
        {
            const float* gp = adj + rowbase + (size_t)t * 4;
            __builtin_amdgcn_global_load_async_to_lds_b128(
                (AS1 v4i*)gp, (AS3 v4i*)&s_adj[0][t * 4], 0, 0);
        }
        for (int c = 0; c < nchunks; ++c) {
            const int buf = c & 1;
            if (c + 1 < nchunks) {
                const float* gp = adj + rowbase + (size_t)(c + 1) * 1024 + t * 4;
                __builtin_amdgcn_global_load_async_to_lds_b128(
                    (AS1 v4i*)gp, (AS3 v4i*)&s_adj[buf ^ 1][t * 4], 0, 0);
                WAIT_ASYNC(1);
            } else {
                WAIT_ASYNC(0);
            }
            float4 a = *(const float4*)&s_adj[buf][t * 4];
            int cb = (c << 10) + (t << 2);
            if (a.x > 0.0f) { int p = atomicAdd(&s_cnt, 1); if (p < NODES_DEG) s_nbr[p] = cb + 0; }
            if (a.y > 0.0f) { int p = atomicAdd(&s_cnt, 1); if (p < NODES_DEG) s_nbr[p] = cb + 1; }
            if (a.z > 0.0f) { int p = atomicAdd(&s_cnt, 1); if (p < NODES_DEG) s_nbr[p] = cb + 2; }
            if (a.w > 0.0f) { int p = atomicAdd(&s_cnt, 1); if (p < NODES_DEG) s_nbr[p] = cb + 3; }
        }
#else
        const float4* arow = (const float4*)(adj + (size_t)i * n);
        const int nvec = n >> 2;
        for (int v = t; v < nvec; v += 256) {
            float4 a = arow[v];
            int cb = v << 2;
            if (a.x > 0.0f) { int p = atomicAdd(&s_cnt, 1); if (p < NODES_DEG) s_nbr[p] = cb + 0; }
            if (a.y > 0.0f) { int p = atomicAdd(&s_cnt, 1); if (p < NODES_DEG) s_nbr[p] = cb + 1; }
            if (a.z > 0.0f) { int p = atomicAdd(&s_cnt, 1); if (p < NODES_DEG) s_nbr[p] = cb + 2; }
            if (a.w > 0.0f) { int p = atomicAdd(&s_cnt, 1); if (p < NODES_DEG) s_nbr[p] = cb + 3; }
        }
#endif
    }
    __syncthreads();
    if (t == 0) {                                 // pad empty slots with self
        int c = s_cnt; if (c > NODES_DEG) c = NODES_DEG;
        for (int r = c; r < NODES_DEG; ++r) s_nbr[r] = i;
    }
    __syncthreads();

    // ---------------- Phase B: candidate pool gather -----------------------
    for (int j = t; j < SORT_N; j += 256) {
        int v;
        if (j < CAND_C) {
            int src = (j < NODES_DEG * NODES_K) ? s_nbr[j >> 4] : i;
            v = cs[src * NODES_K + (j & 15)];
        } else {
            v = 0x7FFFFFFF;                       // sort sentinel
        }
        s_cand[j] = v;
    }

    // ---------------- Phase C: bitonic sort (1024, ascending) --------------
    for (int kk = 2; kk <= SORT_N; kk <<= 1) {
        for (int jj = kk >> 1; jj > 0; jj >>= 1) {
            __syncthreads();
            for (int idx = t; idx < SORT_N; idx += 256) {
                int ixj = idx ^ jj;
                if (ixj > idx) {
                    int a = s_cand[idx], b = s_cand[ixj];
                    bool up = ((idx & kk) == 0);
                    if ((a > b) == up) { s_cand[idx] = b; s_cand[ixj] = a; }
                }
            }
        }
    }
    __syncthreads();

    // ---------------- Phase D: similarities via WMMA -----------------------
    // A: x[i] replicated over 16 rows; B: 16 candidate feature columns.
    // V_WMMA_F32_16X16X32_F16 operand k order used here (contiguous, same for A and B):
    //   A lane<16 : halves = K0..15
    //   A lane>=16: halves = K16..31
    //   B lane<16 : col = lane,    halves = K0..15
    //   B lane>=16: col = lane-16, halves = K16..31
    {
        if (t < 16) s_sim[CAND_C + t] = NEG_DEAD;  // tail never selected

        const int col = lane & 15;
        const int hk  = lane >> 4;                 // 0 or 1 (K half)
        v16h a0, a1;
        #pragma unroll
        for (int h = 0; h < 8; ++h) {
            a0[h]     = s_xi[hk * 16 + h];         // K 0..31 region
            a0[h + 8] = s_xi[hk * 16 + 8 + h];     // K 0..31 region
            a1[h]     = s_xi[32 + hk * 16 + h];    // K 32..63 region
            a1[h + 8] = s_xi[40 + hk * 16 + h];    // K 32..63 region
        }

#if HAVE_TDM
        // TDM gather-mode staging, double buffered per wave. The sorted
        // candidate list satisfies the gather-mode monotonic-index rule.
        int tile = wave;
        int buf = 0;
        if (tile < NTILES) {
            int rows[16];
            #pragma unroll
            for (int r = 0; r < 16; ++r) rows[r] = s_cand[tile * 16 + r];
            tdm_gather_rows(xh, rows,
                            (unsigned)(size_t)&s_feat[wave][0][0][0], n);
        }
        for (; tile < NTILES; tile += 8) {
            int nxt = tile + 8;
            if (nxt < NTILES) {
                int rows[16];
                #pragma unroll
                for (int r = 0; r < 16; ++r) rows[r] = s_cand[nxt * 16 + r];
                tdm_gather_rows(xh, rows,
                                (unsigned)(size_t)&s_feat[wave][buf ^ 1][0][0], n);
                WAIT_TENSOR(1);
            } else {
                WAIT_TENSOR(0);
            }
            const _Float16* fr = &s_feat[wave][buf][col][0];
            v16h b0 = *(const v16h*)(fr + hk * 16);
            v16h b1 = *(const v16h*)(fr + 32 + hk * 16);
            v8f c = {};
            c = __builtin_amdgcn_wmma_f32_16x16x32_f16(false, a0, false, b0,
                                                       (short)0, c, false, false);
            c = __builtin_amdgcn_wmma_f32_16x16x32_f16(false, a1, false, b1,
                                                       (short)0, c, false, false);
            int j = tile * 16 + col;
            if (lane < 16) {                       // D row M=0, N=lane -> c[0]
                bool first = (j == 0) || (s_cand[j] != s_cand[j - 1]);
                s_sim[j] = first ? c[0] : NEG_MASK;
            }
            buf ^= 1;
        }
#else
        for (int tile = wave; tile < NTILES; tile += 8) {
            int j = tile * 16 + col;
            int cnode = s_cand[j];
            const v16h* bp = (const v16h*)(xh + (size_t)cnode * NODES_D);
            v16h b0 = bp[hk];
            v16h b1 = bp[2 + hk];
            v8f c = {};
            c = __builtin_amdgcn_wmma_f32_16x16x32_f16(false, a0, false, b0,
                                                       (short)0, c, false, false);
            c = __builtin_amdgcn_wmma_f32_16x16x32_f16(false, a1, false, b1,
                                                       (short)0, c, false, false);
            if (lane < 16) {
                bool first = (j == 0) || (s_cand[j] != s_cand[j - 1]);
                s_sim[j] = first ? c[0] : NEG_MASK;
            }
        }
#endif
    }
    __syncthreads();

    // ---------------- Phase E1: unique count + ordered unique list ---------
    {
        int cnt = 0;
        for (int j = t; j < CAND_C; j += 256) {
            bool first = (j == 0) || (s_cand[j] != s_cand[j - 1]);
            if (first) cnt++;
        }
        atomicAdd(&s_nu, cnt);
        if (t == 0) {
            int u = 0;
            for (int j = 0; j < CAND_C && u < NODES_K; ++j) {
                bool first = (j == 0) || (s_cand[j] != s_cand[j - 1]);
                if (first) s_uniq[u++] = s_cand[j];
            }
            for (; u < NODES_K; ++u) s_uniq[u] = 0;
        }
    }
    __syncthreads();

    // ---------------- Phase E2: top-K by iterative argmax ------------------
    for (int k = 0; k < NODES_K; ++k) {
        float bv = NEG_DEAD; int bi = 0x3FFFFFFF;
        for (int j = t; j < CAND_C + 16; j += 256) {
            float v = s_sim[j];
            if (v > bv || (v == bv && j < bi)) { bv = v; bi = j; }
        }
        s_rval[t] = bv; s_ridx[t] = bi;
        __syncthreads();
        for (int s = 128; s > 0; s >>= 1) {
            if (t < s) {
                float ov = s_rval[t + s]; int oi = s_ridx[t + s];
                if (ov > s_rval[t] || (ov == s_rval[t] && oi < s_ridx[t])) {
                    s_rval[t] = ov; s_ridx[t] = oi;
                }
            }
            __syncthreads();
        }
        if (t == 0) {
            int b = s_ridx[0];
            s_sel[k] = s_cand[b];
            s_sim[b] = NEG_DEAD;                   // remove winner
        }
        __syncthreads();
    }

    // ---------------- Output (with <K-unique fallback) ---------------------
    if (t < NODES_K) {
        int nu = s_nu;
        int o;
        if (nu >= NODES_K) {
            o = s_sel[t];
        } else if (t < nu) {
            o = s_uniq[t];
        } else {
            int jj = t - nu;
            if (jj > NODES_K - 1) jj = NODES_K - 1;
            if (jj < 0) jj = 0;
            o = cs[i * NODES_K + jj];
        }
        out[i * NODES_K + t] = o;
    }
}

extern "C" void kernel_launch(void* const* d_in, const int* in_sizes, int n_in,
                              void* d_out, int out_size, void* d_ws, size_t ws_size,
                              hipStream_t stream) {
    const float* x   = (const float*)d_in[0];     // [N, 64] f32
    const float* adj = (const float*)d_in[1];     // [N, N]  f32
    const int*   cs  = (const int*)d_in[2];       // [N, 16] i32
    int* out = (int*)d_out;                       // [N, 16] i32

    const int n = in_sizes[0] / NODES_D;          // 8192
    _Float16* xh = (_Float16*)d_ws;               // f16 feature table, 1 MB

    const int total = n * NODES_D;
    attn_cvt_f16_kernel<<<(total + 255) / 256, 256, 0, stream>>>(x, xh, total);
    attn_sample_update_kernel<<<n, 256, 0, stream>>>(adj, xh, cs, out, n);
}
